// MambaEncoder_8040178778270
// MI455X (gfx1250) — matrix operand order repaired and checked by reading, not compiled, not run on hardware
//
#include <hip/hip_runtime.h>
#include <math.h>

typedef __attribute__((ext_vector_type(16))) __bf16 v16bf;
typedef __attribute__((ext_vector_type(8)))  __bf16 v8bf;
typedef __attribute__((ext_vector_type(8)))  float  v8f;

#define B_      2
#define L_      1024
#define DMODEL  512
#define DSTATE  16
#define HEADDIM 64
#define DINNER  1024
#define NHEADS  16
#define DCONV   4
#define NLAYERS 6
#define FFN_    2048
#define CONVDIM 1056              // DINNER + 2*DSTATE
#define DPROJ   2096              // 2*DINNER + 2*DSTATE + NHEADS
#define DTOFF   2080              // DINNER + CONVDIM
#define ROWS    (B_*L_)           // 2048

// Use gfx1250 async global->LDS DMA for A-panel staging (inline asm; flip to 0
// if the assembler rejects the mnemonic).
#define USE_ASYNC_LDS 1

// ---------------- helpers ----------------

__device__ __forceinline__ float blk_reduce256(float v) {
  __shared__ float sm[256];
  int t = threadIdx.x;
  sm[t] = v; __syncthreads();
  for (int s = 128; s > 0; s >>= 1) { if (t < s) sm[t] += sm[t + s]; __syncthreads(); }
  float r = sm[0]; __syncthreads();
  return r;
}

__device__ __forceinline__ float silu(float x) { return x / (1.f + __expf(-x)); }

// ---------------- elementwise / norm kernels ----------------

// pack fp32 B[K][N] (row-major) into bf16 WMMA B-fragment order.
// K-tile = 32 rows, N-tile = 16 cols; tile t = kt*(N/16) + nt.
// out[((t*32 + lane)*16 + j)] = B[kt*32 + (lane>>4)*8 + (j&7) + (j>>3)*16][nt*16 + (lane&15)]
// so each lane's 16 values (K = 8h..8h+7, 16+8h..16+8h+7) are contiguous 32 bytes.
__global__ void k_pack_b(const float* __restrict__ in, __bf16* __restrict__ out,
                         int K, int N) {
  int i = blockIdx.x * blockDim.x + threadIdx.x;
  if (i >= K * N) return;
  int j    = i & 15;
  int lane = (i >> 4) & 31;
  int t    = i >> 9;                 // 512 elements per tile
  int tilesN = N >> 4;
  int kt = t / tilesN, nt = t % tilesN;
  int k = kt * 32 + (lane >> 4) * 8 + (j & 7) + (j >> 3) * 16;
  int n = nt * 16 + (lane & 15);
  out[i] = (__bf16)in[(size_t)k * N + n];
}

// h = rms(x, w, 1e-6); s = (h + pos) * mask; write s (bf16) fwd and time-flipped bwd
__global__ void k_pre_ssm(const float* __restrict__ x, const float* __restrict__ pos,
                          const float* __restrict__ mask, const float* __restrict__ w,
                          __bf16* __restrict__ sf, __bf16* __restrict__ sb) {
  int bl = blockIdx.x;
  int b = bl / L_, l = bl % L_;
  const float* xr = x + (size_t)bl * DMODEL;
  float v0 = xr[threadIdx.x], v1 = xr[threadIdx.x + 256];
  float ss = blk_reduce256(v0 * v0 + v1 * v1);
  float inv = rsqrtf(ss / (float)DMODEL + 1e-6f);
  float m = mask[bl];
  size_t fo = (size_t)bl * DMODEL;
  size_t bo = ((size_t)b * L_ + (L_ - 1 - l)) * DMODEL;
  for (int j = 0; j < 2; ++j) {
    int d = threadIdx.x + j * 256;
    float s = (xr[d] * inv * w[d] + pos[fo + d]) * m;
    sf[fo + d] = (__bf16)s;
    sb[bo + d] = (__bf16)s;
  }
}

// dt = softplus(zx[...,2080+h] + dt_bias[h]); dA = exp(-exp(A_log[h]) * dt)
__global__ void k_dt(const float* __restrict__ zx, const float* __restrict__ dtb,
                     const float* __restrict__ alog, float* __restrict__ dt,
                     float* __restrict__ dA) {
  int i = blockIdx.x * blockDim.x + threadIdx.x;
  if (i >= ROWS * NHEADS) return;
  int h = i & (NHEADS - 1);
  float v = zx[(size_t)(i >> 4) * DPROJ + DTOFF + h] + dtb[h];
  float d = (v > 20.f) ? v : log1pf(__expf(v));
  dt[i] = d;
  dA[i] = __expf(-__expf(alog[h]) * d);
}

// causal depthwise conv (DCONV=4) over zx channels [DINNER, DINNER+CONVDIM) + SiLU
__global__ void k_conv(const float* __restrict__ zx, const float* __restrict__ cw,
                       const float* __restrict__ cb, float* __restrict__ xbc) {
  int i = blockIdx.x * blockDim.x + threadIdx.x;
  if (i >= ROWS * CONVDIM) return;
  int c = i % CONVDIM; int bl = i / CONVDIM;
  int b = bl / L_, l = bl % L_;
  float acc = cb[c];
#pragma unroll
  for (int k = 0; k < DCONV; ++k) {
    int lp = l - (DCONV - 1) + k;
    if (lp >= 0)
      acc += zx[((size_t)b * L_ + lp) * DPROJ + DINNER + c] * cw[c * DCONV + k];
  }
  xbc[i] = silu(acc);
}

// selective-state scan: one block per (b, head), 64 lanes = HEADDIM rows,
// 16-wide state kept in registers; y = scan_y + D[h]*x
__global__ void k_scan(const float* __restrict__ xbc, const float* __restrict__ dt,
                       const float* __restrict__ dA, const float* __restrict__ Dvec,
                       float* __restrict__ y) {
  int b = blockIdx.x >> 4;
  int h = blockIdx.x & 15;
  int p = threadIdx.x;          // 0..63
  float hs[DSTATE];
#pragma unroll
  for (int n = 0; n < DSTATE; ++n) hs[n] = 0.f;
  float Dh = Dvec[h];
  for (int l = 0; l < L_; ++l) {
    size_t base = (size_t)b * L_ + l;
    const float* xr = xbc + base * CONVDIM;
    float dAv = dA[base * NHEADS + h];
    float dtv = dt[base * NHEADS + h];
    float xv  = xr[h * HEADDIM + p];
    float dtx = dtv * xv;
    float yv = 0.f;
#pragma unroll
    for (int n = 0; n < DSTATE; ++n) {
      hs[n] = hs[n] * dAv + dtx * xr[DINNER + n];          // B
      yv   += hs[n] * xr[DINNER + DSTATE + n];             // C
    }
    y[base * DINNER + h * HEADDIM + p] = yv + Dh * xv;
  }
}

// g = y * silu(z); rms(g, mnorm_w, 1e-5) -> bf16
__global__ void k_gate_norm(const float* __restrict__ y, const float* __restrict__ zx,
                            const float* __restrict__ w, __bf16* __restrict__ out) {
  int bl = blockIdx.x;
  const float* yr = y + (size_t)bl * DINNER;
  const float* zr = zx + (size_t)bl * DPROJ;   // z = channels [0,1024)
  float g[4]; float ss = 0.f;
#pragma unroll
  for (int j = 0; j < 4; ++j) {
    int d = threadIdx.x + j * 256;
    float gg = yr[d] * silu(zr[d]);
    g[j] = gg; ss += gg * gg;
  }
  ss = blk_reduce256(ss);
  float inv = rsqrtf(ss / (float)DINNER + 1e-5f);
#pragma unroll
  for (int j = 0; j < 4; ++j) {
    int d = threadIdx.x + j * 256;
    out[(size_t)bl * DINNER + d] = (__bf16)(g[j] * inv * w[d]);
  }
}

// x += 0.5*(o_fwd + flip(o_bwd))*mask
__global__ void k_combine(float* __restrict__ x, const float* __restrict__ of,
                          const float* __restrict__ ob, const float* __restrict__ mask) {
  int i = blockIdx.x * blockDim.x + threadIdx.x;
  if (i >= ROWS * DMODEL) return;
  int d = i % DMODEL; int bl = i / DMODEL;
  int b = bl / L_, l = bl % L_;
  size_t bo = ((size_t)b * L_ + (L_ - 1 - l)) * DMODEL + d;
  x[i] += 0.5f * (of[i] + ob[bo]) * mask[bl];
}

// rms(x, w, 1e-6) -> bf16 (for FFN input)
__global__ void k_rms_bf16(const float* __restrict__ x, const float* __restrict__ w,
                           __bf16* __restrict__ out) {
  int bl = blockIdx.x;
  const float* xr = x + (size_t)bl * DMODEL;
  float v0 = xr[threadIdx.x], v1 = xr[threadIdx.x + 256];
  float ss = blk_reduce256(v0 * v0 + v1 * v1);
  float inv = rsqrtf(ss / (float)DMODEL + 1e-6f);
  for (int j = 0; j < 2; ++j) {
    int d = threadIdx.x + j * 256;
    out[(size_t)bl * DMODEL + d] = (__bf16)(xr[d] * inv * w[d]);
  }
}

__global__ void k_bias_gelu(const float* __restrict__ g, const float* __restrict__ bias,
                            __bf16* __restrict__ out) {
  int i = blockIdx.x * blockDim.x + threadIdx.x;
  if (i >= ROWS * FFN_) return;
  float v = g[i] + bias[i % FFN_];
  out[i] = (__bf16)(0.5f * v * (1.f + erff(v * 0.70710678118f)));
}

__global__ void k_bias_res(float* __restrict__ x, const float* __restrict__ g,
                           const float* __restrict__ bias) {
  int i = blockIdx.x * blockDim.x + threadIdx.x;
  if (i >= ROWS * DMODEL) return;
  x[i] += g[i] + bias[i % DMODEL];
}

__global__ void k_final_rms(const float* __restrict__ x, const float* __restrict__ w,
                            float* __restrict__ out) {
  int bl = blockIdx.x;
  const float* xr = x + (size_t)bl * DMODEL;
  float v0 = xr[threadIdx.x], v1 = xr[threadIdx.x + 256];
  float ss = blk_reduce256(v0 * v0 + v1 * v1);
  float inv = rsqrtf(ss / (float)DMODEL + 1e-6f);
  for (int j = 0; j < 2; ++j) {
    int d = threadIdx.x + j * 256;
    out[(size_t)bl * DMODEL + d] = xr[d] * inv * w[d];
  }
}

// ---------------- WMMA GEMM (LDS-staged A, ping-pong double-buffered) ----------------
// C[M,N] (f32) = A[M,K] (bf16 row-major) @ B (bf16, pre-packed fragment order)
// Block = 8 wave32s covering one 64-row M strip x 8 consecutive 16-col N tiles.
// A panels staged into LDS (4 KB / k-tile, double buffered) via the gfx1250
// async global->LDS DMA (ASYNCcnt) when USE_ASYNC_LDS, else via VGPR copies.
// All 4 A fragments load as one ds_load_b128 burst (sched_barrier keeps the
// backend from re-interleaving), then 4 v_wmma_f32_16x16x32_bf16 issue
// back-to-back.  Manual 2x k-unroll ping-pongs (smA[0],b0)/(smA[1],b1).

typedef __attribute__((address_space(3))) __bf16 lds_bf16;

__device__ __forceinline__ void stage_chunk(const __bf16* gsrc, __bf16* ldst) {
#if USE_ASYNC_LDS
  // LDS byte offset for the VDST operand of the async DMA
  unsigned lofs = (unsigned)(size_t)(lds_bf16*)ldst;
  asm volatile("global_load_async_to_lds_b128 %0, %1, off"
               :: "v"(lofs), "v"(gsrc) : "memory");
#else
  *(v8bf*)ldst = *(const v8bf*)gsrc;
#endif
}

__device__ __forceinline__ void stage_drain() {
#if USE_ASYNC_LDS
  asm volatile("s_wait_asynccnt 0" ::: "memory");
#endif
}

__device__ __forceinline__ void wmma_tile(const __bf16* __restrict__ buf,
                                          const v16bf& bfr, v8f acc[4],
                                          int r16, int kh) {
  v16bf afr[4];
#pragma unroll
  for (int mi = 0; mi < 4; ++mi) {
    const __bf16* lp = buf + (mi * 16 + r16) * 32 + kh * 8;
    v8bf alo = *(const v8bf*)lp;          // K = kh*8 .. +7
    v8bf ahi = *(const v8bf*)(lp + 16);   // K = 16+kh*8 .. +7
    afr[mi] = __builtin_shufflevector(alo, ahi, 0, 1, 2, 3, 4, 5, 6, 7,
                                      8, 9, 10, 11, 12, 13, 14, 15);
  }
#if __has_builtin(__builtin_amdgcn_sched_barrier)
  __builtin_amdgcn_sched_barrier(0);   // keep the 8 ds_loads as one burst
#endif
#pragma unroll
  for (int mi = 0; mi < 4; ++mi)
    acc[mi] = __builtin_amdgcn_wmma_f32_16x16x32_bf16(false, afr[mi], false, bfr,
                                                      (short)0, acc[mi], false, false);
}

__global__ void k_gemm_bf16(const __bf16* __restrict__ A, const __bf16* __restrict__ Bp,
                            float* __restrict__ C, int M, int N, int K) {
  __shared__ __align__(16) __bf16 smA[2][64 * 32];
  const int tilesN  = N >> 4;
  const int nblocks = (tilesN + 7) >> 3;
  const int mb = blockIdx.x / nblocks;
  const int nb = blockIdx.x % nblocks;
  const int m0 = mb << 6;
  const int lane = threadIdx.x & 31;
  const int wave = threadIdx.x >> 5;
  const int nt = nb * 8 + wave;
  const bool valid = (nt < tilesN);          // wave-uniform
  const int r16 = lane & 15;                  // A row within 16-tile / C column
  const int kh  = lane >> 4;                  // lane-half selects K sub-range
  const int ktiles = K >> 5;                  // always even here (16/32/64)

  // A-staging assignment: thread -> one 16B chunk of the 64x32 panel
  const int srow = threadIdx.x >> 2;          // 0..63
  const int scol = (threadIdx.x & 3) << 3;    // 0,8,16,24
  const __bf16* Astage = A + (size_t)(m0 + srow) * K + scol;
  __bf16* Lstage0 = &smA[0][srow * 32 + scol];
  __bf16* Lstage1 = &smA[1][srow * 32 + scol];

  // prologue: stage k-tile 0, load B(0)
  stage_chunk(Astage, Lstage0);
  v8f acc[4] = {};
  v16bf b0 = {}, b1 = {};
  if (valid) b0 = *(const v16bf*)(Bp + ((size_t)nt * 32 + lane) * 16);
  stage_drain();
  __syncthreads();

  for (int kt = 0; kt < ktiles; kt += 2) {
    // ---- even tile kt: compute (smA[0], b0); stage (smA[1], b1) for kt+1 ----
    stage_chunk(Astage + ((kt + 1) << 5), Lstage1);
    if (valid) {
      b1 = *(const v16bf*)(Bp + ((size_t)((kt + 1) * tilesN + nt) * 32 + lane) * 16);
      int pf = (kt + 2 < ktiles) ? kt + 2 : ktiles - 1;   // clamped, branchless
      __builtin_prefetch(Bp + ((size_t)(pf * tilesN + nt) * 32 + lane) * 16, 0, 1);
      wmma_tile(smA[0], b0, acc, r16, kh);
    }
    stage_drain();
    __syncthreads();

    // ---- odd tile kt+1: compute (smA[1], b1); stage (smA[0], b0) for kt+2 ----
    if (kt + 2 < ktiles) {
      stage_chunk(Astage + ((kt + 2) << 5), Lstage0);
      if (valid)
        b0 = *(const v16bf*)(Bp + ((size_t)((kt + 2) * tilesN + nt) * 32 + lane) * 16);
    }
    if (valid) wmma_tile(smA[1], b1, acc, r16, kh);
    stage_drain();
    __syncthreads();
  }

  if (valid) {
    const int n0 = nt << 4;
#pragma unroll
    for (int mi = 0; mi < 4; ++mi) {
      float* Cp = C + (size_t)(m0 + mi * 16 + kh * 8) * N + n0 + r16;
#pragma unroll
      for (int r = 0; r < 8; ++r) Cp[(size_t)r * N] = acc[mi][r];
    }
  }
}

// ---------------- host orchestration ----------------

static inline void launch_gemm(const __bf16* A, const __bf16* Bp, float* C,
                               int M, int N, int K, hipStream_t s) {
  int tilesN = N / 16;
  int blocks = (M / 64) * ((tilesN + 7) / 8);
  k_gemm_bf16<<<blocks, 256, 0, s>>>(A, Bp, C, M, N, K);
}

static inline void launch_pack(const float* in, __bf16* out, int K, int N, hipStream_t s) {
  int n = K * N;
  k_pack_b<<<(n + 255) / 256, 256, 0, s>>>(in, out, K, N);
}

extern "C" void kernel_launch(void* const* d_in, const int* in_sizes, int n_in,
                              void* d_out, int out_size, void* d_ws, size_t ws_size,
                              hipStream_t stream) {
  (void)in_sizes; (void)n_in; (void)out_size; (void)ws_size;
  const float* x_in  = (const float*)d_in[0];
  const float* pos   = (const float*)d_in[1];
  const float* mask  = (const float*)d_in[2];
  const float* W_in  = (const float*)d_in[3];   // (6,2,512,2096)
  const float* convw = (const float*)d_in[4];   // (6,2,1056,4)
  const float* convb = (const float*)d_in[5];   // (6,2,1056)
  const float* A_log = (const float*)d_in[6];   // (6,2,16)
  const float* Dvec  = (const float*)d_in[7];   // (6,2,16)
  const float* dtb   = (const float*)d_in[8];   // (6,2,16)
  const float* mnw   = (const float*)d_in[9];   // (6,2,1024)
  const float* W_out = (const float*)d_in[10];  // (6,2,1024,512)
  const float* nssm  = (const float*)d_in[11];  // (6,512)
  const float* fw1   = (const float*)d_in[12];  // (6,512,2048)
  const float* fb1   = (const float*)d_in[13];  // (6,2048)
  const float* fw2   = (const float*)d_in[14];  // (6,2048,512)
  const float* fb2   = (const float*)d_in[15];  // (6,512)
  const float* nffn  = (const float*)d_in[16];  // (6,512)
  const float* fnw   = (const float*)d_in[17];  // (512)

  char* p = (char*)d_ws;
  auto alloc = [&](size_t bytes) -> char* {
    char* r = p; p += (bytes + 255) & ~(size_t)255; return r;
  };
  float*  xbuf  = (float*) alloc((size_t)ROWS * DMODEL * 4);
  float*  o_f   = (float*) alloc((size_t)ROWS * DMODEL * 4);
  float*  o_b   = (float*) alloc((size_t)ROWS * DMODEL * 4);
  __bf16* s_f   = (__bf16*)alloc((size_t)ROWS * DMODEL * 2);
  __bf16* s_b   = (__bf16*)alloc((size_t)ROWS * DMODEL * 2);
  float*  zx    = (float*) alloc((size_t)ROWS * DPROJ * 4);
  float*  xbc   = (float*) alloc((size_t)ROWS * CONVDIM * 4);
  float*  dt_b  = (float*) alloc((size_t)ROWS * NHEADS * 4);
  float*  dA_b  = (float*) alloc((size_t)ROWS * NHEADS * 4);
  float*  ybuf  = (float*) alloc((size_t)ROWS * DINNER * 4);
  __bf16* agate = (__bf16*)alloc((size_t)ROWS * DINNER * 2);
  __bf16* hf    = (__bf16*)alloc((size_t)ROWS * DMODEL * 2);
  float*  g1    = (float*) alloc((size_t)ROWS * FFN_ * 4);
  __bf16* tbf   = (__bf16*)alloc((size_t)ROWS * FFN_ * 2);
  float*  g2    = (float*) alloc((size_t)ROWS * DMODEL * 4);
  __bf16* win   = (__bf16*)alloc((size_t)DMODEL * DPROJ * 2);
  __bf16* wout  = (__bf16*)alloc((size_t)DINNER * DMODEL * 2);
  __bf16* w1b   = (__bf16*)alloc((size_t)DMODEL * FFN_ * 2);
  __bf16* w2b   = (__bf16*)alloc((size_t)FFN_ * DMODEL * 2);

  hipMemcpyAsync(xbuf, x_in, (size_t)ROWS * DMODEL * 4, hipMemcpyDeviceToDevice, stream);

  for (int i = 0; i < NLAYERS; ++i) {
    // ---- pre-SSM norm + pos + mask, produce fwd and flipped bwd inputs ----
    k_pre_ssm<<<ROWS, 256, 0, stream>>>(xbuf, pos, mask, nssm + (size_t)i * DMODEL, s_f, s_b);

    for (int d = 0; d < 2; ++d) {
      size_t sl = (size_t)(i * 2 + d);
      const __bf16* s_in = d ? s_b : s_f;
      float* o_out = d ? o_b : o_f;

      // in-projection (weights packed to fragment order on the fly)
      launch_pack(W_in + sl * DMODEL * DPROJ, win, DMODEL, DPROJ, stream);
      launch_gemm(s_in, win, zx, ROWS, DPROJ, DMODEL, stream);

      // dt / dA / conv+silu
      k_dt<<<(ROWS * NHEADS + 255) / 256, 256, 0, stream>>>(
          zx, dtb + sl * NHEADS, A_log + sl * NHEADS, dt_b, dA_b);
      k_conv<<<(ROWS * CONVDIM + 255) / 256, 256, 0, stream>>>(
          zx, convw + sl * CONVDIM * DCONV, convb + sl * CONVDIM, xbc);

      // sequential selective scan
      k_scan<<<B_ * NHEADS, HEADDIM, 0, stream>>>(xbc, dt_b, dA_b, Dvec + sl * NHEADS, ybuf);

      // gate + group norm -> bf16, out-projection
      k_gate_norm<<<ROWS, 256, 0, stream>>>(ybuf, zx, mnw + sl * DINNER, agate);
      launch_pack(W_out + sl * DINNER * DMODEL, wout, DINNER, DMODEL, stream);
      launch_gemm(agate, wout, o_out, ROWS, DMODEL, DINNER, stream);
    }

    // residual merge (bwd stream un-flipped here)
    k_combine<<<(ROWS * DMODEL + 255) / 256, 256, 0, stream>>>(xbuf, o_f, o_b, mask);

    // ---- FFN ----
    k_rms_bf16<<<ROWS, 256, 0, stream>>>(xbuf, nffn + (size_t)i * DMODEL, hf);
    launch_pack(fw1 + (size_t)i * DMODEL * FFN_, w1b, DMODEL, FFN_, stream);
    launch_gemm(hf, w1b, g1, ROWS, FFN_, DMODEL, stream);
    k_bias_gelu<<<(ROWS * FFN_ + 255) / 256, 256, 0, stream>>>(g1, fb1 + (size_t)i * FFN_, tbf);
    launch_pack(fw2 + (size_t)i * FFN_ * DMODEL, w2b, FFN_, DMODEL, stream);
    launch_gemm(tbf, w2b, g2, ROWS, DMODEL, FFN_, stream);
    k_bias_res<<<(ROWS * DMODEL + 255) / 256, 256, 0, stream>>>(xbuf, g2, fb2 + (size_t)i * DMODEL);
  }

  k_final_rms<<<ROWS, 256, 0, stream>>>(xbuf, fnw, (float*)d_out);
}
